// Block_21715354649057
// MI455X (gfx1250) — matrix operand order repaired and checked by reading, not compiled, not run on hardware
//
#include <hip/hip_runtime.h>
#include <hip/hip_bf16.h>
#include <math.h>

// ---------------------------------------------------------------------------
// CDNA5 (gfx1250, wave32) WMMA helpers
// ---------------------------------------------------------------------------
typedef __attribute__((ext_vector_type(16))) __bf16       v16bf;
typedef __attribute__((ext_vector_type(8)))  float        v8f;
typedef __attribute__((ext_vector_type(4)))  unsigned int v4u;

union Frag16 { v4u q[2]; v16bf v; };

// A-operand (16x32 bf16, row-major MxK source): lane row = lane&15,
// half = lane>>4. VGPR i<4 holds K = 8*half + 2i,+1 ; i>=4 holds K = 16+8*half+2(i-4).
// => two contiguous 16-byte loads at +0 and +32 bytes from (row, k0 + 8*half).
__device__ __forceinline__ v16bf ld_a_frag(const __bf16* rowp, int k0, int half) {
  Frag16 f;
  const v4u* p = (const v4u*)(rowp + k0 + 8 * half);
  f.q[0] = p[0];
  f.q[1] = p[2];
  return f.v;
}

// B-operand (32x16 bf16) supplied as Bt row-major NxK: same K order as A,
// => two contiguous 16-byte loads at +0 and +32 bytes from (row, k0 + 8*half).
__device__ __forceinline__ v16bf ld_b_frag(const __bf16* rowp, int k0, int half) {
  Frag16 f;
  const v4u* p = (const v4u*)(rowp + k0 + 8 * half);
  f.q[0] = p[0];
  f.q[1] = p[2];
  return f.v;
}

__device__ __forceinline__ v8f wmma_bf16(v16bf a, v16bf b, v8f c) {
  return __builtin_amdgcn_wmma_f32_16x16x32_bf16(false, a, false, b, (short)0, c,
                                                 false, false);
}

__device__ __forceinline__ float gelu_f(float x) {
  float x3 = x * x * x;
  return 0.5f * x * (1.0f + tanhf(0.7978845608028654f * (x + 0.044715f * x3)));
}

// ---------------------------------------------------------------------------
// Weight convert + transpose through LDS (coalesced both sides):
// W[k][n] (f32) -> Wt[n][k] (bf16).  grid = (N/32, K/32), block = 256.
// ---------------------------------------------------------------------------
__global__ void k_wt_cvt(const float* __restrict__ W, __bf16* __restrict__ Wt,
                         int K, int N) {
  __shared__ __bf16 tile[32][33];
  int nt = blockIdx.x * 32, kt = blockIdx.y * 32;
  int tx = threadIdx.x & 31, ty = threadIdx.x >> 5;  // 8 rows per pass
#pragma unroll
  for (int r = 0; r < 32; r += 8)
    tile[ty + r][tx] = (__bf16)W[(size_t)(kt + ty + r) * N + nt + tx];
  __syncthreads();
#pragma unroll
  for (int r = 0; r < 32; r += 8)
    Wt[(size_t)(nt + ty + r) * K + kt + tx] = tile[tx][ty + r];
}

// ---------------------------------------------------------------------------
// RMSNorm over rows of 1024, f32 in -> bf16 out
// ---------------------------------------------------------------------------
__global__ void k_rmsnorm(const float* __restrict__ X, const float* __restrict__ w,
                          __bf16* __restrict__ Y) {
  __shared__ float red[256];
  int row = blockIdx.x;
  int tid = threadIdx.x;
  const float* xr = X + (size_t)row * 1024;
  float v0 = xr[tid], v1 = xr[tid + 256], v2 = xr[tid + 512], v3 = xr[tid + 768];
  red[tid] = v0 * v0 + v1 * v1 + v2 * v2 + v3 * v3;
  __syncthreads();
  for (int s = 128; s > 0; s >>= 1) {
    if (tid < s) red[tid] += red[tid + s];
    __syncthreads();
  }
  float scale = rsqrtf(red[0] * (1.0f / 1024.0f) + 1e-6f);
  __bf16* yr = Y + (size_t)row * 1024;
  yr[tid]       = (__bf16)(v0 * scale * w[tid]);
  yr[tid + 256] = (__bf16)(v1 * scale * w[tid + 256]);
  yr[tid + 512] = (__bf16)(v2 * scale * w[tid + 512]);
  yr[tid + 768] = (__bf16)(v3 * scale * w[tid + 768]);
}

// ---------------------------------------------------------------------------
// bf16 WMMA GEMM, compile-time N/K so that all tile strides become 24-bit
// instruction immediates (two live base pointers, no spills):
//   C[M,N] = alpha*(A @ B) (+bias)(->gelu)(+residual)
// A row-major MxK bf16, Bt row-major NxK bf16 (i.e. B column-major).
// Block: 256 thr = 8 waves. Register tile per wave: 2x4 WMMA tiles (32x64).
// Block tile: 128x128.  grid = (N/128, M/128).
// Per k-step: 12 x b128 loads feed 8 WMMAs (A frags reused 4x, B frags 2x).
// ---------------------------------------------------------------------------
template <int N, int K>
__global__ void k_gemm(const __bf16* __restrict__ A, const __bf16* __restrict__ Bt,
                       float alpha, const float* __restrict__ bias,
                       const float* __restrict__ residual, int use_gelu,
                       float* __restrict__ Cf, __bf16* __restrict__ Cb) {
  int wave = threadIdx.x >> 5, lane = threadIdx.x & 31;
  int lane16 = lane & 15, half = lane >> 4;
  int m0 = blockIdx.y * 128 + (wave >> 1) * 32;  // row tiles m0, m0+16
  int n0 = blockIdx.x * 128 + (wave & 1) * 64;   // col tiles n0 + j*16
  const __bf16* arow = A + (size_t)(m0 + lane16) * K;   // +16*K for 2nd tile (imm)
  const __bf16* brow = Bt + (size_t)(n0 + lane16) * K;  // +j*16*K (imm)
  v8f acc[2][4] = {};
  for (int k0 = 0; k0 < K; k0 += 32) {
    v16bf a0 = ld_a_frag(arow, k0, half);
    v16bf a1 = ld_a_frag(arow + 16 * K, k0, half);
#pragma unroll
    for (int j = 0; j < 4; ++j) {
      v16bf bfrag = ld_b_frag(brow + j * 16 * K, k0, half);
      __builtin_prefetch(brow + j * 16 * K + k0 + 1024, 0, 1);
      acc[0][j] = wmma_bf16(a0, bfrag, acc[0][j]);
      acc[1][j] = wmma_bf16(a1, bfrag, acc[1][j]);
    }
  }
#pragma unroll
  for (int t = 0; t < 2; ++t) {
#pragma unroll
    for (int j = 0; j < 4; ++j) {
      int n = n0 + j * 16 + lane16;
#pragma unroll
      for (int r = 0; r < 8; ++r) {
        int m = m0 + t * 16 + r + 8 * half;
        size_t idx = (size_t)m * N + n;
        float v = acc[t][j][r] * alpha;
        if (bias) v += bias[n];
        if (use_gelu) v = gelu_f(v);
        if (residual) v += residual[idx];
        if (Cf) Cf[idx] = v;
        if (Cb) Cb[idx] = (__bf16)v;
      }
    }
  }
}

// ---------------------------------------------------------------------------
// Chunkwise retention, split over value-columns for parallelism.
// grid = (H=8, B=4, EG=4), block = 256 (8 waves). Each block owns a 32-wide
// e-slice of the 128x128 state (the recurrence is independent per e-column).
// Chunk C=64:
//   Am = mask_decay(Q K^T)                      (64x64, recomputed per slice)
//   Y[:,eslice] = Am @ V[:,eslice] + diag(g^{i+1}) Q @ S_prev[:,eslice]
//   S[:,eslice] = g^64 * S_prev[:,eslice] + (decay-weighted K)^T @ V[:,eslice]
// State master f32 in LDS transposed St[e][d]; bf16 shadow for WMMA B-operand.
// Dynamic LDS: 86016 B -> multiple workgroups per WGP fit (<320KB).
// ---------------------------------------------------------------------------
__global__ void k_retnet(const __bf16* __restrict__ Qb, const __bf16* __restrict__ Kb,
                         const __bf16* __restrict__ Vb, const float* __restrict__ Sin,
                         __bf16* __restrict__ Yb, float* __restrict__ Sout) {
  extern __shared__ char smem[];
  float*  S_f = (float*)smem;                      // 32*128 f32  (16K)
  __bf16* Stb = (__bf16*)(smem + 16384);           // 32*128 bf16 (8K)
  __bf16* Qs  = (__bf16*)(smem + 24576);           // 64*128      (16K)
  __bf16* Ks  = (__bf16*)(smem + 40960);           // 64*128      (16K)
  __bf16* KhT = (__bf16*)(smem + 57344);           // 128*64      (16K)
  __bf16* Vt  = (__bf16*)(smem + 73728);           // 32*64       (4K)
  __bf16* Am  = (__bf16*)(smem + 77824);           // 64*64       (8K)

  const int h = blockIdx.x, b = blockIdx.y, eg = blockIdx.z;
  const int e0 = eg * 32;
  const int tid = threadIdx.x;
  const int wave = tid >> 5, lane = tid & 31, lane16 = lane & 15, half = lane >> 4;
  const float gamma = 1.0f - exp2f(-5.0f - (float)h);
  const float lg = log2f(gamma);
  const float gC = exp2f(64.0f * lg);
  const size_t sbase = ((size_t)(b * 8 + h)) * 16384;

  // init state slice transposed: S_f[e*128+d] = Sin[d][e0+e]
  for (int idx = tid; idx < 32 * 128; idx += 256) {
    int e = idx >> 7, d = idx & 127;
    S_f[e * 128 + d] = Sin[sbase + (size_t)d * 128 + e0 + e];
  }
  __syncthreads();

  for (int c = 0; c < 64; ++c) {
    const int t0 = c * 64;
    // phase 0: stage chunk Q/K (full d) + decayed K^T + V slice + bf16 state
    for (int idx = tid; idx < 64 * 128; idx += 256) {
      int t = idx >> 7, d = idx & 127;
      size_t g = ((size_t)(b * 4096 + t0 + t)) * 1024 + h * 128 + d;
      __bf16 qv = Qb[g], kv = Kb[g];
      Qs[idx] = qv;
      Ks[idx] = kv;
      KhT[d * 64 + t] = (__bf16)((float)kv * exp2f((float)(63 - t) * lg));
    }
    for (int idx = tid; idx < 32 * 64; idx += 256) {
      int e = idx >> 6, t = idx & 63;
      size_t g = ((size_t)(b * 4096 + t0 + t)) * 1024 + h * 128 + e0 + e;
      Vt[e * 64 + t] = Vb[g];
    }
    for (int idx = tid; idx < 32 * 128; idx += 256) Stb[idx] = (__bf16)S_f[idx];
    __syncthreads();

    // phase 1: Am = decay-masked Q @ K^T   (16 tiles, 2 per wave)
#pragma unroll
    for (int tt = 0; tt < 2; ++tt) {
      int tile = wave + tt * 8;
      int ti = tile >> 2, tj = tile & 3;
      const __bf16* arow = Qs + (ti * 16 + lane16) * 128;
      const __bf16* brow = Ks + (tj * 16 + lane16) * 128;
      v8f acc = {};
#pragma unroll
      for (int k0 = 0; k0 < 128; k0 += 32)
        acc = wmma_bf16(ld_a_frag(arow, k0, half), ld_b_frag(brow, k0, half), acc);
#pragma unroll
      for (int r = 0; r < 8; ++r) {
        int i = ti * 16 + r + 8 * half;
        int j = tj * 16 + lane16;
        float v = (j <= i) ? acc[r] * exp2f((float)(i - j) * lg) : 0.0f;
        Am[i * 64 + j] = (__bf16)v;
      }
    }
    __syncthreads();

    // phase 2a: Y tiles (8 tiles, 1 per wave): intra + cross
    {
      int ti = wave >> 1, td = wave & 1;
      const __bf16* ai_a = Am + (ti * 16 + lane16) * 64;
      const __bf16* ai_b = Vt + (td * 16 + lane16) * 64;
      v8f acc_i = {};
      acc_i = wmma_bf16(ld_a_frag(ai_a, 0, half), ld_b_frag(ai_b, 0, half), acc_i);
      acc_i = wmma_bf16(ld_a_frag(ai_a, 32, half), ld_b_frag(ai_b, 32, half), acc_i);
      const __bf16* ac_a = Qs + (ti * 16 + lane16) * 128;
      const __bf16* ac_b = Stb + (td * 16 + lane16) * 128;
      v8f acc_c = {};
#pragma unroll
      for (int k0 = 0; k0 < 128; k0 += 32)
        acc_c = wmma_bf16(ld_a_frag(ac_a, k0, half), ld_b_frag(ac_b, k0, half), acc_c);
#pragma unroll
      for (int r = 0; r < 8; ++r) {
        int i = ti * 16 + r + 8 * half;
        float y = acc_i[r] + exp2f((float)(i + 1) * lg) * acc_c[r];
        size_t g = ((size_t)(b * 4096 + t0 + i)) * 1024 + h * 128 + e0 + td * 16 + lane16;
        Yb[g] = (__bf16)y;
      }
    }
    // phase 2b: state update (16 tiles, 2 per wave), stored transposed
#pragma unroll
    for (int s = 0; s < 2; ++s) {
      const __bf16* arow = KhT + (wave * 16 + lane16) * 64;  // rows = d
      const __bf16* brow = Vt + (s * 16 + lane16) * 64;      // rows = e (slice)
      v8f acc = {};
      acc = wmma_bf16(ld_a_frag(arow, 0, half), ld_b_frag(brow, 0, half), acc);
      acc = wmma_bf16(ld_a_frag(arow, 32, half), ld_b_frag(brow, 32, half), acc);
#pragma unroll
      for (int r = 0; r < 8; ++r) {
        int d = wave * 16 + r + 8 * half;
        int e = s * 16 + lane16;
        S_f[e * 128 + d] = gC * S_f[e * 128 + d] + acc[r];
      }
    }
    __syncthreads();
  }

  for (int idx = tid; idx < 32 * 128; idx += 256) {
    int e = idx >> 7, d = idx & 127;
    Sout[sbase + (size_t)d * 128 + e0 + e] = S_f[e * 128 + d];
  }
}

// ---------------------------------------------------------------------------
// Host orchestration
// ---------------------------------------------------------------------------
extern "C" void kernel_launch(void* const* d_in, const int* in_sizes, int n_in,
                              void* d_out, int out_size, void* d_ws, size_t ws_size,
                              hipStream_t stream) {
  (void)in_sizes; (void)n_in; (void)out_size; (void)ws_size;
  const float* x   = (const float*)d_in[0];
  const float* st0 = (const float*)d_in[1];
  const float* n1w = (const float*)d_in[2];
  const float* wq  = (const float*)d_in[3];
  const float* wk  = (const float*)d_in[4];
  const float* wv  = (const float*)d_in[5];
  const float* wo  = (const float*)d_in[6];
  const float* n2w = (const float*)d_in[7];
  const float* w1  = (const float*)d_in[8];
  const float* b1  = (const float*)d_in[9];
  const float* w2  = (const float*)d_in[10];
  const float* b2  = (const float*)d_in[11];

  float* out_x = (float*)d_out;
  float* out_s = out_x + (size_t)16384 * 1024;

  char* ws = (char*)d_ws;
  const size_t MB = 1024 * 1024;
  __bf16* wqT = (__bf16*)(ws + 0 * MB);
  __bf16* wkT = (__bf16*)(ws + 2 * MB);
  __bf16* wvT = (__bf16*)(ws + 4 * MB);
  __bf16* woT = (__bf16*)(ws + 6 * MB);
  __bf16* w1T = (__bf16*)(ws + 8 * MB);    // [4096][1024]
  __bf16* w2T = (__bf16*)(ws + 16 * MB);   // [1024][4096]
  __bf16* xn  = (__bf16*)(ws + 24 * MB);   // 32MB
  __bf16* qb  = (__bf16*)(ws + 56 * MB);   // 32MB
  __bf16* kb  = (__bf16*)(ws + 88 * MB);   // 32MB
  __bf16* vb  = (__bf16*)(ws + 120 * MB);  // 32MB
  __bf16* yb  = (__bf16*)(ws + 152 * MB);  // 32MB
  float*  x1  = (float*)(ws + 184 * MB);   // 64MB ; total 248MB
  __bf16* hb  = (__bf16*)(ws + 152 * MB);  // alias yb (dead after wo GEMM)
  __bf16* h1  = (__bf16*)(ws + 24 * MB);   // alias xn/q/k/v (dead after retention)

  const int M = 16384;

  k_wt_cvt<<<dim3(32, 32), 256, 0, stream>>>(wq, wqT, 1024, 1024);
  k_wt_cvt<<<dim3(32, 32), 256, 0, stream>>>(wk, wkT, 1024, 1024);
  k_wt_cvt<<<dim3(32, 32), 256, 0, stream>>>(wv, wvT, 1024, 1024);
  k_wt_cvt<<<dim3(32, 32), 256, 0, stream>>>(wo, woT, 1024, 1024);
  k_wt_cvt<<<dim3(128, 32), 256, 0, stream>>>(w1, w1T, 1024, 4096);
  k_wt_cvt<<<dim3(32, 128), 256, 0, stream>>>(w2, w2T, 4096, 1024);

  k_rmsnorm<<<M, 256, 0, stream>>>(x, n1w, xn);

  const float qscale = 0.08838834764831845f;  // D^-0.5, D=128
  k_gemm<1024, 1024><<<dim3(8, M / 128), 256, 0, stream>>>(
      xn, wqT, qscale, nullptr, nullptr, 0, nullptr, qb);
  k_gemm<1024, 1024><<<dim3(8, M / 128), 256, 0, stream>>>(
      xn, wkT, 1.0f, nullptr, nullptr, 0, nullptr, kb);
  k_gemm<1024, 1024><<<dim3(8, M / 128), 256, 0, stream>>>(
      xn, wvT, 1.0f, nullptr, nullptr, 0, nullptr, vb);

  k_retnet<<<dim3(8, 4, 4), 256, 86016, stream>>>(qb, kb, vb, st0, yb, out_s);

  // x1 = y @ wo + skip(x)
  k_gemm<1024, 1024><<<dim3(8, M / 128), 256, 0, stream>>>(
      yb, woT, 1.0f, nullptr, x, 0, x1, nullptr);
  k_rmsnorm<<<M, 256, 0, stream>>>(x1, n2w, hb);
  // h1 = gelu(h @ w1 + b1)
  k_gemm<4096, 1024><<<dim3(32, M / 128), 256, 0, stream>>>(
      hb, w1T, 1.0f, b1, nullptr, 1, nullptr, h1);
  // out = h1 @ w2 + b2 + x1
  k_gemm<1024, 4096><<<dim3(8, M / 128), 256, 0, stream>>>(
      h1, w2T, 1.0f, b2, x1, 0, out_x, nullptr);
}
